// VanishingNet_test_83820581749513
// MI455X (gfx1250) — matrix operand order repaired and checked by reading, not compiled, not run on hardware
//
#include <hip/hip_runtime.h>
#include <hip/hip_bf16.h>
#include <stdint.h>

#define N_TOTAL 32768
#define N0      8192
#define N1      1024
#define KNB     8
#define C_IN    128
#define C_H     64
#define COS_T   0.999f
#define NEGV    -1e30f

typedef _Float16 half_t;
typedef __attribute__((ext_vector_type(16))) _Float16 v16h;
typedef __attribute__((ext_vector_type(8)))  _Float16 v8h;
typedef __attribute__((ext_vector_type(8)))  float    v8f;

// ---------------------------------------------------------------------------
// Conversion / transpose kernels (run once per launch; everything stays in L2)
// ---------------------------------------------------------------------------
__global__ void feat_transpose_half(const float* __restrict__ feat,
                                    half_t* __restrict__ featT) {
  int i = blockIdx.x * blockDim.x + threadIdx.x;  // i = k * N_TOTAL + n (coalesced read)
  if (i >= C_IN * N_TOTAL) return;
  int k = i >> 15;
  int n = i & (N_TOTAL - 1);
  featT[(size_t)n * C_IN + k] = (half_t)feat[i];
}

__global__ void f32_to_half(const float* __restrict__ src, half_t* __restrict__ dst, int n) {
  int i = blockIdx.x * blockDim.x + threadIdx.x;
  if (i < n) dst[i] = (half_t)src[i];
}

// W: [4][64][64] (row = K, col = N).  WT[layer][n][k] = W[layer][k][n]
// -> B-fragment loads become contiguous 8-K v8h loads.
__global__ void dgcn_w_to_half_T(const float* __restrict__ W, half_t* __restrict__ WT) {
  int i = blockIdx.x * blockDim.x + threadIdx.x;
  if (i >= 4 * C_H * C_H) return;
  int layer = i >> 12;
  int nn = (i >> 6) & 63;
  int kk = i & 63;
  WT[i] = (half_t)W[(layer << 12) + (kk << 6) + nn];
}

// ---------------------------------------------------------------------------
// WMMA fragment helpers (CDNA5 16x16x32 f16 layouts, CDNA5 ISA)
// A 16x32 f16: lane l<16 -> row M=l, K = {0..7} in v0..3, {16..23} in v4..7;
//              lane l>=16 -> row M=l-16, K offsets +8 / +24.
// B 32x16 f16: lane l<16 -> col N=l, K={0..7},{16..23} (2 per VGPR); lanes >=16 K offsets +8 / +24.
// C/D 16x16 f32: lane<16 -> N=lane, VGPR r -> M=r; lane>=16 -> M=r+8.
// ---------------------------------------------------------------------------
__device__ __forceinline__ v16h load_a16x32(const half_t* rowp, int koff) {
  union { v16h v; v8h h[2]; } u;
  u.h[0] = *(const v8h*)(rowp + koff);        // K = koff .. koff+7
  u.h[1] = *(const v8h*)(rowp + koff + 16);   // K = koff+16 .. koff+23
  return u.v;
}

__device__ __forceinline__ v16h load_b32x16(const half_t* WT, int ldw,
                                            int n0, int k0, int l, int hi) {
  // WT is [N][K] row-major: lane's K = k0+8*hi+{0..7} and k0+16+8*hi+{0..7}.
  union { v16h v; v8h h[2]; } u;
  const half_t* p = WT + (size_t)(n0 + l) * ldw + k0 + hi * 8;
  u.h[0] = *(const v8h*)(p);
  u.h[1] = *(const v8h*)(p + 16);
  return u.v;
}

// ---------------------------------------------------------------------------
// sphere_feat:  X[s, 0:64] = featT[ind[s], 0:128] @ W^T + b   (M=S, K=128, N=64)
// one wave per 16-row tile, 4 N-tiles -> 16 WMMAs per wave
// ---------------------------------------------------------------------------
__global__ void __launch_bounds__(256) sphere_feat_wmma(
    const half_t* __restrict__ featT, const int* __restrict__ ind,
    const half_t* __restrict__ Wh, const float* __restrict__ bias,
    half_t* __restrict__ Xout, int S) {
  int wave = threadIdx.x >> 5;
  int m0 = (blockIdx.x * 8 + wave) * 16;
  if (m0 >= S) return;
  int lane = threadIdx.x & 31;
  int l = lane & 15, hi = lane >> 4;
  const half_t* arow = featT + (size_t)ind[m0 + l] * C_IN;
  v8f acc[4] = {};
  for (int k0 = 0; k0 < C_IN; k0 += 32) {
    v16h a = load_a16x32(arow, k0 + hi * 8);
#pragma unroll
    for (int t = 0; t < 4; ++t) {
      v16h b = load_b32x16(Wh, C_IN, t * 16, k0, l, hi);
      acc[t] = __builtin_amdgcn_wmma_f32_16x16x32_f16(false, a, false, b,
                                                      (short)0, acc[t], false, false);
    }
  }
#pragma unroll
  for (int t = 0; t < 4; ++t) {
    int nn = t * 16 + l;
    float bv = bias[nn];
    int mbase = m0 + hi * 8;
#pragma unroll
    for (int r = 0; r < 8; ++r)
      Xout[(size_t)(mbase + r) * C_H + nn] = (half_t)(acc[t][r] + bv);
  }
}

// ---------------------------------------------------------------------------
// neighbor mean aggregation: Agg[n,c] = (1/8) sum_k X[nbr[n*8+k], c]
// ---------------------------------------------------------------------------
__global__ void agg_kernel(const half_t* __restrict__ X, const int* __restrict__ nbr,
                           half_t* __restrict__ Agg, int n) {
  int i = blockIdx.x * blockDim.x + threadIdx.x;  // i = node*64 + c
  if (i >= n * C_H) return;
  int node = i >> 6, c = i & 63;
  const int* nb = nbr + node * KNB;
  float s = 0.f;
#pragma unroll
  for (int k = 0; k < KNB; ++k) s += (float)X[(size_t)nb[k] * C_H + c];
  Agg[i] = (half_t)(s * 0.125f);
}

// ---------------------------------------------------------------------------
// dgcn layer: Xout = relu(X @ Wself + Agg @ Wnb + bias)   (M=n, K=64, N=64)
// two WMMA accumulation chains into the same accumulators
// ---------------------------------------------------------------------------
__global__ void __launch_bounds__(256) dgcn_layer_wmma(
    const half_t* __restrict__ X, const half_t* __restrict__ Agg,
    const half_t* __restrict__ WsT, const half_t* __restrict__ WnT,
    const float* __restrict__ bias, half_t* __restrict__ Xout, int n) {
  int wave = threadIdx.x >> 5;
  int m0 = (blockIdx.x * 8 + wave) * 16;
  if (m0 >= n) return;
  int lane = threadIdx.x & 31;
  int l = lane & 15, hi = lane >> 4;
  const half_t* xrow = X + (size_t)(m0 + l) * C_H;
  const half_t* grow = Agg + (size_t)(m0 + l) * C_H;
  v8f acc[4] = {};
#pragma unroll
  for (int k0 = 0; k0 < C_H; k0 += 32) {
    v16h a = load_a16x32(xrow, k0 + hi * 8);
#pragma unroll
    for (int t = 0; t < 4; ++t) {
      v16h b = load_b32x16(WsT, C_H, t * 16, k0, l, hi);
      acc[t] = __builtin_amdgcn_wmma_f32_16x16x32_f16(false, a, false, b,
                                                      (short)0, acc[t], false, false);
    }
  }
#pragma unroll
  for (int k0 = 0; k0 < C_H; k0 += 32) {
    v16h a = load_a16x32(grow, k0 + hi * 8);
#pragma unroll
    for (int t = 0; t < 4; ++t) {
      v16h b = load_b32x16(WnT, C_H, t * 16, k0, l, hi);
      acc[t] = __builtin_amdgcn_wmma_f32_16x16x32_f16(false, a, false, b,
                                                      (short)0, acc[t], false, false);
    }
  }
#pragma unroll
  for (int t = 0; t < 4; ++t) {
    int nn = t * 16 + l;
    float bv = bias[nn];
    int mbase = m0 + hi * 8;
#pragma unroll
    for (int r = 0; r < 8; ++r) {
      float v = acc[t][r] + bv;
      Xout[(size_t)(mbase + r) * C_H + nn] = (half_t)(v > 0.f ? v : 0.f);
    }
  }
}

// ---------------------------------------------------------------------------
// head: s[n] = X[n,:] . head + hb ; also write sigmoid(s) to output slice
// ---------------------------------------------------------------------------
__global__ void head_kernel(const half_t* __restrict__ X, const float* __restrict__ head,
                            const float* __restrict__ hb, float* __restrict__ scores,
                            float* __restrict__ outSig, int n) {
  int node = blockIdx.x * blockDim.x + threadIdx.x;
  if (node >= n) return;
  const half_t* xr = X + (size_t)node * C_H;
  float s = hb[0];
#pragma unroll
  for (int c = 0; c < C_H; ++c) s += (float)xr[c] * head[c];
  scores[node] = s;
  outSig[node] = 1.f / (1.f + expf(-s));
}

// ---------------------------------------------------------------------------
// block argmax helper (wave32 shuffle + LDS), first-occurrence tie-break
// ---------------------------------------------------------------------------
__device__ __forceinline__ void block_argmax(float v, int idx, float* swv, int* swi,
                                             float* outv, int* outi) {
  __syncthreads();
#pragma unroll
  for (int off = 16; off; off >>= 1) {
    float ov = __shfl_xor(v, off);
    int   oi = __shfl_xor(idx, off);
    if (ov > v || (ov == v && oi < idx)) { v = ov; idx = oi; }
  }
  int lane = threadIdx.x & 31, w = threadIdx.x >> 5;
  if (lane == 0) { swv[w] = v; swi[w] = idx; }
  __syncthreads();
  if (threadIdx.x == 0) {
    int nw = blockDim.x >> 5;
    float bv = swv[0]; int bi = swi[0];
    for (int i = 1; i < nw; ++i)
      if (swv[i] > bv || (swv[i] == bv && swi[i] < bi)) { bv = swv[i]; bi = swi[i]; }
    *outv = bv; *outi = bi;
  }
  __syncthreads();
}

// ---------------------------------------------------------------------------
// greedy NMS-3 over stage-0 scores with antipodal angular suppression
// ---------------------------------------------------------------------------
__global__ void __launch_bounds__(1024) nms3_kernel(
    const float* __restrict__ scores, const float* __restrict__ xyz,
    const int* __restrict__ ind0, int* __restrict__ anchors) {
  __shared__ float s[N0];
  __shared__ float swv[32]; __shared__ int swi[32];
  __shared__ float bv; __shared__ int bi;
  __shared__ float px, py, pz;
  int t = threadIdx.x;
  for (int i = t; i < N0; i += 1024) s[i] = scores[i];
  __syncthreads();
  for (int iter = 0; iter < 3; ++iter) {
    float v = -3.4e38f; int vi = 0x7fffffff;
    for (int i = t; i < N0; i += 1024)
      if (s[i] > v || (s[i] == v && i < vi)) { v = s[i]; vi = i; }
    block_argmax(v, vi, swv, swi, &bv, &bi);
    if (t == 0) {
      int g = ind0[bi];
      anchors[iter] = g;
      px = xyz[g * 3 + 0]; py = xyz[g * 3 + 1]; pz = xyz[g * 3 + 2];
    }
    __syncthreads();
    for (int i = t; i < N0; i += 1024) {
      int g = ind0[i];
      float d = fabsf(xyz[g * 3] * px + xyz[g * 3 + 1] * py + xyz[g * 3 + 2] * pz);
      if (d >= COS_T) s[i] = NEGV;
    }
    __syncthreads();
  }
}

// per-branch argmax -> anchor = ind[b*N1 + argmax]
__global__ void __launch_bounds__(1024) branch_argmax_kernel(
    const float* __restrict__ scores, const int* __restrict__ ind, int* __restrict__ anchors) {
  __shared__ float swv[32]; __shared__ int swi[32];
  __shared__ float bv; __shared__ int bi;
  for (int b = 0; b < 3; ++b) {
    float v = scores[b * N1 + threadIdx.x];
    block_argmax(v, (int)threadIdx.x, swv, swi, &bv, &bi);
    if (threadIdx.x == 0) anchors[b] = ind[b * N1 + bi];
    __syncthreads();
  }
}

// ind_out[b*N1+j] = ind_scale[anchors[b]*N1 + j]
__global__ void gather_ind_kernel(const int* __restrict__ ind_scale,
                                  const int* __restrict__ anchors, int* __restrict__ ind_out) {
  int i = blockIdx.x * blockDim.x + threadIdx.x;
  if (i >= 3 * N1) return;
  ind_out[i] = ind_scale[(size_t)anchors[i >> 10] * N1 + (i & (N1 - 1))];
}

__global__ void gather_xyz_kernel(const float* __restrict__ xyz, const int* __restrict__ ind,
                                  float* __restrict__ P) {
  int i = blockIdx.x * blockDim.x + threadIdx.x;
  if (i >= 3 * N1) return;
  int g = ind[i];
  P[i * 3 + 0] = xyz[g * 3 + 0];
  P[i * 3 + 1] = xyz[g * 3 + 1];
  P[i * 3 + 2] = xyz[g * 3 + 2];
}

// per row: top-8 of |dot| within the row's branch (stable ties -> earlier idx)
__global__ void topk8_kernel(const float* __restrict__ P, int* __restrict__ nbr) {
  int r = blockIdx.x * blockDim.x + threadIdx.x;
  if (r >= 3 * N1) return;
  int base = (r >> 10) << 10;
  float x = P[3 * r], y = P[3 * r + 1], z = P[3 * r + 2];
  float v[KNB]; int id[KNB];
#pragma unroll
  for (int k = 0; k < KNB; ++k) { v[k] = -1.0f; id[k] = 0; }
  for (int m = 0; m < N1; ++m) {
    const float* q = P + 3 * (base + m);
    float d = fabsf(x * q[0] + y * q[1] + z * q[2]);
    if (d > v[KNB - 1]) {
      int pos = KNB - 1;
#pragma unroll
      for (int k = KNB - 1; k > 0; --k)
        if (d > v[k - 1]) { v[k] = v[k - 1]; id[k] = id[k - 1]; pos = k - 1; }
      v[pos] = d; id[pos] = base + m;     // global node index within 3*N1
    }
  }
#pragma unroll
  for (int k = 0; k < KNB; ++k) nbr[r * KNB + k] = id[k];
}

// ---------------------------------------------------------------------------
extern "C" void kernel_launch(void* const* d_in, const int* in_sizes, int n_in,
                              void* d_out, int out_size, void* d_ws, size_t ws_size,
                              hipStream_t stream) {
  (void)in_sizes; (void)n_in; (void)out_size; (void)ws_size;

  const float* feat  = (const float*)d_in[0];
  const float* xyz   = (const float*)d_in[1];
  const float* wsF[3]  = {(const float*)d_in[2], (const float*)d_in[4], (const float*)d_in[6]};
  const float* bsF[3]  = {(const float*)d_in[3], (const float*)d_in[5], (const float*)d_in[7]};
  const float* WselfF[3] = {(const float*)d_in[8],  (const float*)d_in[13], (const float*)d_in[18]};
  const float* WnbF[3]   = {(const float*)d_in[9],  (const float*)d_in[14], (const float*)d_in[19]};
  const float* dgbF[3]   = {(const float*)d_in[10], (const float*)d_in[15], (const float*)d_in[20]};
  const float* headF[3]  = {(const float*)d_in[11], (const float*)d_in[16], (const float*)d_in[21]};
  const float* hbF[3]    = {(const float*)d_in[12], (const float*)d_in[17], (const float*)d_in[22]};
  const int* edge0 = (const int*)d_in[23];
  const int* ind0  = (const int*)d_in[24];
  const int* ind1s = (const int*)d_in[25];
  const int* ind2s = (const int*)d_in[26];
  const int* nbr0  = edge0 + N0 * KNB;   // edge0 row 1 = neighbor ids

  float* out = (float*)d_out;            // [8192 | 3072 | 3072]

  // -------- workspace carve-out (256B aligned slots) --------
  uint8_t* wsb = (uint8_t*)d_ws;
  size_t off = 0;
  auto alloc = [&](size_t bytes) -> void* {
    void* p = wsb + off;
    off = (off + bytes + 255) & ~(size_t)255;
    return p;
  };
  half_t* featT   = (half_t*)alloc((size_t)N_TOTAL * C_IN * 2);
  half_t* wsphH[3]; for (int i = 0; i < 3; ++i) wsphH[i]  = (half_t*)alloc(C_H * C_IN * 2);
  half_t* wselfT[3]; for (int i = 0; i < 3; ++i) wselfT[i] = (half_t*)alloc(4 * C_H * C_H * 2);
  half_t* wnbT[3];   for (int i = 0; i < 3; ++i) wnbT[i]   = (half_t*)alloc(4 * C_H * C_H * 2);
  half_t* Xa      = (half_t*)alloc((size_t)N0 * C_H * 2);
  half_t* Xb      = (half_t*)alloc((size_t)N0 * C_H * 2);
  half_t* AggB    = (half_t*)alloc((size_t)N0 * C_H * 2);
  float*  scoresB = (float*)alloc(N0 * 4);
  int* anchors1 = (int*)alloc(256);
  int* anchors2 = (int*)alloc(256);
  int* indBuf1  = (int*)alloc(3 * N1 * 4);
  int* indBuf2  = (int*)alloc(3 * N1 * 4);
  float* Pbuf   = (float*)alloc(3 * N1 * 3 * 4);
  int* nbrBuf   = (int*)alloc(3 * N1 * KNB * 4);

  // -------- one-time conversions --------
  feat_transpose_half<<<(C_IN * N_TOTAL) / 256, 256, 0, stream>>>(feat, featT);
  for (int i = 0; i < 3; ++i) {
    f32_to_half<<<(C_H * C_IN + 255) / 256, 256, 0, stream>>>(wsF[i], wsphH[i], C_H * C_IN);
    dgcn_w_to_half_T<<<(4 * C_H * C_H) / 256, 256, 0, stream>>>(WselfF[i], wselfT[i]);
    dgcn_w_to_half_T<<<(4 * C_H * C_H) / 256, 256, 0, stream>>>(WnbF[i],   wnbT[i]);
  }

  // -------- dgcn runner (4 layers, ping-pong) --------
  auto run_dgcn = [&](const int* nbr, const half_t* WsT, const half_t* WnT,
                      const float* biasAll, int n) -> half_t* {
    half_t* cur = Xa; half_t* nxt = Xb;
    for (int i = 0; i < 4; ++i) {
      agg_kernel<<<(n * C_H) / 256, 256, 0, stream>>>(cur, nbr, AggB, n);
      dgcn_layer_wmma<<<n / 128, 256, 0, stream>>>(cur, AggB, WsT + i * C_H * C_H,
                                                   WnT + i * C_H * C_H, biasAll + i * C_H,
                                                   nxt, n);
      half_t* t = cur; cur = nxt; nxt = t;
    }
    return cur;
  };

  // -------- stage 0 --------
  sphere_feat_wmma<<<N0 / 128, 256, 0, stream>>>(featT, ind0, wsphH[0], bsF[0], Xa, N0);
  half_t* X0 = run_dgcn(nbr0, wselfT[0], wnbT[0], dgbF[0], N0);
  head_kernel<<<N0 / 256, 256, 0, stream>>>(X0, headF[0], hbF[0], scoresB, out, N0);
  nms3_kernel<<<1, 1024, 0, stream>>>(scoresB, xyz, ind0, anchors1);

  // -------- branch stage runner (stages 1 & 2) --------
  auto run_branch = [&](const int* indScale, const int* anchIn, int* indBuf, int stage,
                        float* outSig, int* anchOut) {
    gather_ind_kernel<<<12, 256, 0, stream>>>(indScale, anchIn, indBuf);
    gather_xyz_kernel<<<12, 256, 0, stream>>>(xyz, indBuf, Pbuf);
    topk8_kernel<<<12, 256, 0, stream>>>(Pbuf, nbrBuf);
    sphere_feat_wmma<<<(3 * N1) / 128, 256, 0, stream>>>(featT, indBuf, wsphH[stage],
                                                         bsF[stage], Xa, 3 * N1);
    half_t* Xf = run_dgcn(nbrBuf, wselfT[stage], wnbT[stage], dgbF[stage], 3 * N1);
    head_kernel<<<(3 * N1) / 256, 256, 0, stream>>>(Xf, headF[stage], hbF[stage],
                                                    scoresB, outSig, 3 * N1);
    if (anchOut)
      branch_argmax_kernel<<<1, 1024, 0, stream>>>(scoresB, indBuf, anchOut);
  };

  run_branch(ind1s, anchors1, indBuf1, 1, out + N0,          anchors2);
  run_branch(ind2s, anchors2, indBuf2, 2, out + N0 + 3 * N1, nullptr);
}
